// edge_aggregation_18562848653356
// MI455X (gfx1250) — matrix operand order repaired and checked by reading, not compiled, not run on hardware
//
#include <hip/hip_runtime.h>
#include <hip/hip_bf16.h>

typedef __bf16 bf16;
typedef __attribute__((ext_vector_type(16))) __bf16 v16bf;
typedef __attribute__((ext_vector_type(8)))  __bf16 bf16x8;
typedef __attribute__((ext_vector_type(8)))  float  v8f;

#define T_EDGE 5
#define D_DIM  256
#define HID_DIM 128
#define B_DIM  4
#define E_DIM  8000
#define N_DIM  2000

__device__ __forceinline__ v8f v8f_splat(float x) {
    v8f r;
#pragma unroll
    for (int j = 0; j < 8; ++j) r[j] = x;
    return r;
}

__device__ __forceinline__ v16bf mkfrag(bf16x8 c0, bf16x8 c1) {
    v16bf f;
#pragma unroll
    for (int j = 0; j < 8; ++j) { f[j] = c0[j]; f[8 + j] = c1[j]; }
    return f;
}

// ---------------------------------------------------------------------------
// K0: convert + transpose weights to bf16.
//   W1t[t][h][d] = W1[t][d][h]   (B-columns of GEMM1 (K=D) contiguous)
//   W2t[t][d][h] = W2[t][h][d]   (B-columns of GEMM2 (K=HID) contiguous)
// ---------------------------------------------------------------------------
__global__ void k0_prep(const float* __restrict__ W1, const float* __restrict__ W2,
                        bf16* __restrict__ W1t, bf16* __restrict__ W2t) {
    int idx = blockIdx.x * blockDim.x + threadIdx.x;
    const int n1 = T_EDGE * D_DIM * HID_DIM;  // 163840
    if (idx < n1) {
        int t = idx / (D_DIM * HID_DIM);
        int r = idx % (D_DIM * HID_DIM);
        int h = r / D_DIM, d = r % D_DIM;
        W1t[idx] = (bf16)W1[(size_t)t * D_DIM * HID_DIM + (size_t)d * HID_DIM + h];
        int d2 = r / HID_DIM, h2 = r % HID_DIM;
        W2t[idx] = (bf16)W2[(size_t)t * HID_DIM * D_DIM + (size_t)h2 * D_DIM + d2];
    }
}

// ---------------------------------------------------------------------------
// K1: edges(bf16)[b,e,d] = H[b,e,:] @ ori[b,:,d].
// 256 threads (8 wave32), tile 64(E) x 256(D); wave = 16 rows x 128 cols.
// Ping-pong LDS: stage tile ks+1 while wmma-ing tile ks (overlaps the HBM
// stream of H with the matrix pipe; H is the 22us roofline term).
// ---------------------------------------------------------------------------
__global__ __launch_bounds__(256)
void k1_edges(const float* __restrict__ H, const float* __restrict__ ori,
              bf16* __restrict__ edgesbf) {
    __shared__ __align__(16) bf16 ldsA[2][64 * 32];    // H tile, row-major
    __shared__ __align__(16) bf16 ldsB[2][256 * 32];   // ori tile, transposed [n][k]

    const int b = blockIdx.y;
    const int e0 = blockIdx.x * 64;
    const int tid = threadIdx.x;
    const int w = tid >> 5, lane = tid & 31;
    const int rg = w & 3, ch = w >> 2;              // row-group / column-half
    const int col = lane & 15, hi = lane >> 4;
    const int kbA = hi * 8;     // A-frag: K chunks at kbA and kbA+16
    const int kbB = hi * 8;     // B-frag: K chunks at kbB and kbB+16

    const float* Hb = H + (size_t)b * E_DIM * N_DIM;
    const float* orib = ori + (size_t)b * N_DIM * D_DIM;

    auto stage = [&](int ks, int p) {
        const int k0 = ks * 32;
#pragma unroll
        for (int i = 0; i < 8; ++i) {                // A: 64x32, coalesced in k
            int idx = tid + i * 256;
            int r = idx >> 5, k = idx & 31;
            float v = (k0 + k < N_DIM) ? Hb[(size_t)(e0 + r) * N_DIM + k0 + k] : 0.f;
            ldsA[p][r * 32 + k] = (bf16)v;
        }
#pragma unroll
        for (int i = 0; i < 32; ++i) {               // B: 32x256 -> [n][k]
            int idx = tid + i * 256;
            int k = idx >> 8, n = idx & 255;
            float v = (k0 + k < N_DIM) ? orib[(size_t)(k0 + k) * D_DIM + n] : 0.f;
            ldsB[p][n * 32 + k] = (bf16)v;
        }
    };

    v8f acc[8];
#pragma unroll
    for (int i = 0; i < 8; ++i) acc[i] = v8f_splat(0.f);

    const int KSTEPS = (N_DIM + 31) / 32;  // 63 (last padded)
    stage(0, 0);
    __syncthreads();

    for (int ks = 0; ks < KSTEPS; ++ks) {
        const int p = ks & 1;
        if (ks + 1 < KSTEPS) stage(ks + 1, p ^ 1);   // overlap next tile with wmma
        if (ks + 2 < KSTEPS && tid < 64)
            __builtin_prefetch(&Hb[(size_t)(e0 + tid) * N_DIM + (ks + 2) * 32], 0, 0);

        const int row = rg * 16 + col;
        v16bf afrag = mkfrag(*(const bf16x8*)&ldsA[p][row * 32 + kbA],
                             *(const bf16x8*)&ldsA[p][row * 32 + kbA + 16]);
        v16bf bfrag[8];
#pragma unroll
        for (int nt = 0; nt < 8; ++nt) {
            const int n = ch * 128 + nt * 16 + col;
            bfrag[nt] = mkfrag(*(const bf16x8*)&ldsB[p][n * 32 + kbB],
                               *(const bf16x8*)&ldsB[p][n * 32 + kbB + 16]);
        }
#pragma unroll
        for (int nt = 0; nt < 8; ++nt)
            acc[nt] = __builtin_amdgcn_wmma_f32_16x16x32_bf16(
                false, afrag, false, bfrag[nt], (short)0, acc[nt], false, false);
        __syncthreads();   // stores of p^1 done + reads of p done
    }

    bf16* outp = edgesbf + (size_t)b * E_DIM * D_DIM;
#pragma unroll
    for (int nt = 0; nt < 8; ++nt)
#pragma unroll
        for (int j = 0; j < 8; ++j) {
            int row = e0 + rg * 16 + j + hi * 8;    // C layout: M = j + 8*(lane>>4)
            outp[(size_t)row * D_DIM + ch * 128 + nt * 16 + col] = (bf16)acc[nt][j];
        }
}

// ---------------------------------------------------------------------------
// K2: edge_feature[b,e,:] = sum_t ed[b,e,t] * (relu(edges@W1[t]+b1[t])@W2[t]+b2[t])
// 8 waves: wave w handles rows 16*(w&3), out-column-half 128*(w>>2).
// Both column-half waves recompute h (compute is free vs the BW roofline);
// each wave transposes its h tile through a private 4KB LDS slice.
// All operands are L2-resident; no intra-loop barriers besides the h hand-off.
// ---------------------------------------------------------------------------
__global__ __launch_bounds__(256)
void k2_mlp(const bf16* __restrict__ edgesbf,
            const bf16* __restrict__ W1t, const bf16* __restrict__ W2t,
            const float* __restrict__ b1, const float* __restrict__ b2,
            const float* __restrict__ ed, float* __restrict__ edge_feat) {
    __shared__ __align__(16) bf16 ldsH[8][16 * 128];

    const int b = blockIdx.y;
    const int e0 = blockIdx.x * 64;
    const int tid = threadIdx.x;
    const int w = tid >> 5, lane = tid & 31;
    const int rg = w & 3, ch = w >> 2;
    const int col = lane & 15, hi = lane >> 4;
    const int kbA = hi * 8, kbB = hi * 8;
    const int erow0 = e0 + rg * 16;

    const bf16* Eb = edgesbf + (size_t)b * E_DIM * D_DIM;

    v8f outacc[8];
#pragma unroll
    for (int i = 0; i < 8; ++i) outacc[i] = v8f_splat(0.f);

    int rows[8];
#pragma unroll
    for (int j = 0; j < 8; ++j) rows[j] = erow0 + j + hi * 8;

    for (int t = 0; t < T_EDGE; ++t) {
        // ---- h = relu(edges @ W1[t] + b1[t]) : [16 x 128]
        v8f hacc[8];
#pragma unroll
        for (int i = 0; i < 8; ++i) hacc[i] = v8f_splat(b1[t * HID_DIM + i * 16 + col]);
#pragma unroll
        for (int ks = 0; ks < 8; ++ks) {          // K = D = 256
            const bf16* ap = Eb + (size_t)(erow0 + col) * D_DIM + ks * 32 + kbA;
            v16bf afrag = mkfrag(*(const bf16x8*)ap, *(const bf16x8*)(ap + 16));
            v16bf bfrag[8];
#pragma unroll
            for (int nt = 0; nt < 8; ++nt) {
                const bf16* bp = W1t + ((size_t)t * HID_DIM + nt * 16 + col) * D_DIM
                                 + ks * 32 + kbB;
                bfrag[nt] = mkfrag(*(const bf16x8*)bp, *(const bf16x8*)(bp + 16));
            }
#pragma unroll
            for (int nt = 0; nt < 8; ++nt)
                hacc[nt] = __builtin_amdgcn_wmma_f32_16x16x32_bf16(
                    false, afrag, false, bfrag[nt], (short)0, hacc[nt], false, false);
        }
        // relu, drop to bf16, transpose via private LDS slice (C layout -> A layout)
#pragma unroll
        for (int nt = 0; nt < 8; ++nt)
#pragma unroll
            for (int j = 0; j < 8; ++j) {
                float v = hacc[nt][j];
                v = v > 0.f ? v : 0.f;
                ldsH[w][(j + hi * 8) * 128 + nt * 16 + col] = (bf16)v;
            }
        __syncthreads();

        // ---- mlp2 = h @ W2[t] + b2[t] : this wave's 16 x 128 output slab
        v8f tacc[8];
#pragma unroll
        for (int i = 0; i < 8; ++i)
            tacc[i] = v8f_splat(b2[t * D_DIM + ch * 128 + i * 16 + col]);
#pragma unroll
        for (int ks = 0; ks < 4; ++ks) {          // K = HID = 128
            const bf16* ap = &ldsH[w][col * 128 + ks * 32 + kbA];
            v16bf afrag = mkfrag(*(const bf16x8*)ap, *(const bf16x8*)(ap + 16));
            v16bf bfrag[8];
#pragma unroll
            for (int nt = 0; nt < 8; ++nt) {
                const bf16* bp = W2t + ((size_t)t * D_DIM + ch * 128 + nt * 16 + col)
                                 * HID_DIM + ks * 32 + kbB;
                bfrag[nt] = mkfrag(*(const bf16x8*)bp, *(const bf16x8*)(bp + 16));
            }
#pragma unroll
            for (int nt = 0; nt < 8; ++nt)
                tacc[nt] = __builtin_amdgcn_wmma_f32_16x16x32_bf16(
                    false, afrag, false, bfrag[nt], (short)0, tacc[nt], false, false);
        }
        // gated accumulate
        float g[8];
#pragma unroll
        for (int j = 0; j < 8; ++j)
            g[j] = ed[((size_t)b * E_DIM + rows[j]) * T_EDGE + t];
#pragma unroll
        for (int nt = 0; nt < 8; ++nt)
#pragma unroll
            for (int j = 0; j < 8; ++j) outacc[nt][j] += g[j] * tacc[nt][j];
        __syncthreads();  // before next t reuses ldsH
    }

    float* op = edge_feat + (size_t)b * E_DIM * D_DIM;
#pragma unroll
    for (int nt = 0; nt < 8; ++nt)
#pragma unroll
        for (int j = 0; j < 8; ++j)
            op[(size_t)rows[j] * D_DIM + ch * 128 + nt * 16 + col] = outacc[nt][j];
}

// ---------------------------------------------------------------------------
// K3: out[b,n,0:256] = H^T @ edge_feature ; out[b,n,256:512] = ori[b,n,:]
// Tile 64(N) x 256(D); 8 waves, wave tile 16 x 128; K = E in steps of 32.
// Same ping-pong staging as K1 (second pass over H).
// ---------------------------------------------------------------------------
__global__ __launch_bounds__(256)
void k3_node(const float* __restrict__ H, const float* __restrict__ ori,
             const float* __restrict__ edge_feat, float* __restrict__ out) {
    __shared__ __align__(16) bf16 ldsA[2][64 * 32];    // H^T tile [n][e]
    __shared__ __align__(16) bf16 ldsB[2][256 * 32];   // edge_feat tile [d][e]

    const int b = blockIdx.y;
    const int n0 = blockIdx.x * 64;
    const int tid = threadIdx.x;
    const int w = tid >> 5, lane = tid & 31;
    const int rg = w & 3, ch = w >> 2;
    const int col = lane & 15, hi = lane >> 4;
    const int kbA = hi * 8, kbB = hi * 8;

    const float* Hb = H + (size_t)b * E_DIM * N_DIM;
    const float* efb = edge_feat + (size_t)b * E_DIM * D_DIM;

    auto stage = [&](int ks, int p) {
        const int k0 = ks * 32;
#pragma unroll
        for (int i = 0; i < 8; ++i) {                // A^T: H[e][n] -> [n][e]
            int idx = tid + i * 256;
            int nl = idx & 63, el = idx >> 6;        // coalesced in n
            float v = (n0 + nl < N_DIM) ? Hb[(size_t)(k0 + el) * N_DIM + n0 + nl] : 0.f;
            ldsA[p][nl * 32 + el] = (bf16)v;
        }
#pragma unroll
        for (int i = 0; i < 32; ++i) {               // B^T: ef[e][d] -> [d][e]
            int idx = tid + i * 256;
            int k = idx >> 8, d = idx & 255;
            ldsB[p][d * 32 + k] = (bf16)efb[(size_t)(k0 + k) * D_DIM + d];
        }
    };

    v8f acc[8];
#pragma unroll
    for (int i = 0; i < 8; ++i) acc[i] = v8f_splat(0.f);

    const int KSTEPS = E_DIM / 32;  // 250
    stage(0, 0);
    __syncthreads();

    for (int ks = 0; ks < KSTEPS; ++ks) {
        const int p = ks & 1;
        if (ks + 1 < KSTEPS) stage(ks + 1, p ^ 1);
        if (ks + 2 < KSTEPS && tid < 32)
            __builtin_prefetch(&Hb[(size_t)((ks + 2) * 32 + tid) * N_DIM + n0], 0, 0);

        const int row = rg * 16 + col;
        v16bf afrag = mkfrag(*(const bf16x8*)&ldsA[p][row * 32 + kbA],
                             *(const bf16x8*)&ldsA[p][row * 32 + kbA + 16]);
        v16bf bfrag[8];
#pragma unroll
        for (int nt = 0; nt < 8; ++nt) {
            const int n = ch * 128 + nt * 16 + col;
            bfrag[nt] = mkfrag(*(const bf16x8*)&ldsB[p][n * 32 + kbB],
                               *(const bf16x8*)&ldsB[p][n * 32 + kbB + 16]);
        }
#pragma unroll
        for (int nt = 0; nt < 8; ++nt)
            acc[nt] = __builtin_amdgcn_wmma_f32_16x16x32_bf16(
                false, afrag, false, bfrag[nt], (short)0, acc[nt], false, false);
        __syncthreads();
    }

    float* ob = out + (size_t)b * N_DIM * (2 * D_DIM);
#pragma unroll
    for (int nt = 0; nt < 8; ++nt)
#pragma unroll
        for (int j = 0; j < 8; ++j) {
            int n = n0 + rg * 16 + j + hi * 8;
            if (n < N_DIM)
                ob[(size_t)n * 512 + ch * 128 + nt * 16 + col] = acc[nt][j];
        }
    // concat ori passthrough
    const float* orib = ori + (size_t)b * N_DIM * D_DIM;
#pragma unroll 4
    for (int i = 0; i < 64; ++i) {
        int idx = tid + i * 256;                     // 0..16383
        int nl = idx >> 8, d = idx & 255;
        int n = n0 + nl;
        if (n < N_DIM) ob[(size_t)n * 512 + 256 + d] = orib[(size_t)n * 256 + d];
    }
}

// ---------------------------------------------------------------------------
extern "C" void kernel_launch(void* const* d_in, const int* in_sizes, int n_in,
                              void* d_out, int out_size, void* d_ws, size_t ws_size,
                              hipStream_t stream) {
    (void)in_sizes; (void)n_in; (void)out_size; (void)ws_size;
    const float* ed  = (const float*)d_in[0];   // [B,E,T]
    const float* H   = (const float*)d_in[1];   // [B,E,N]
    const float* ori = (const float*)d_in[2];   // [B,N,D]
    const float* W1  = (const float*)d_in[3];   // [T,D,HID]
    const float* b1  = (const float*)d_in[4];   // [T,HID]
    const float* W2  = (const float*)d_in[5];   // [T,HID,D]
    const float* b2  = (const float*)d_in[6];   // [T,D]
    float* out = (float*)d_out;                 // [B,N,2D]

    const size_t wsz = (size_t)T_EDGE * D_DIM * HID_DIM;       // 163840
    bf16* W1t = (bf16*)d_ws;
    bf16* W2t = W1t + wsz;
    bf16* edgesbf = W2t + wsz;                                  // [B,E,D] bf16
    float* edge_feat = (float*)(edgesbf + (size_t)B_DIM * E_DIM * D_DIM);  // fp32

    k0_prep<<<dim3((int)((wsz + 255) / 256)), dim3(256), 0, stream>>>(W1, W2, W1t, W2t);
    k1_edges<<<dim3(E_DIM / 64, B_DIM), dim3(256), 0, stream>>>(H, ori, edgesbf);
    k2_mlp<<<dim3(E_DIM / 64, B_DIM), dim3(256), 0, stream>>>(edgesbf, W1t, W2t,
                                                              b1, b2, ed, edge_feat);
    k3_node<<<dim3((N_DIM + 63) / 64, B_DIM), dim3(256), 0, stream>>>(H, ori,
                                                                      edge_feat, out);
}
